// MlaAttentionRef_55078660604220
// MI455X (gfx1250) — matrix operand order repaired and checked by reading, not compiled, not run on hardware
//
#include <hip/hip_runtime.h>
#include <hip/hip_bf16.h>
#include <math.h>

typedef __attribute__((ext_vector_type(16))) __bf16 v16bf;
typedef __attribute__((ext_vector_type(8)))  float  v8f;

namespace {
constexpr int Bb = 2, Ss = 2048, Hh = 16, DNn = 128, DRr = 64, DVv = 128;
// 1/sqrt(192) * log2(e): folded into Q so logits come out pre-scaled in base-2 units
constexpr float QSCALE = 0.07216878364870323f * 1.4426950408889634f;
}

#define WMMA_BF16(A, Bm, C) \
  __builtin_amdgcn_wmma_f32_16x16x32_bf16(false, (A), false, (Bm), (short)0, (C), false, false)

__global__ __launch_bounds__(32) void mla_fwd_kernel(
    const float* __restrict__ qn, const float* __restrict__ qp,
    const float* __restrict__ kn, const float* __restrict__ kp,
    const float* __restrict__ vv, float* __restrict__ o, float* __restrict__ lse)
{
  const int lane = threadIdx.x & 31;
  const int qt = blockIdx.x, h = blockIdx.y, b = blockIdx.z;
  const int qbase = qt * 32;                 // 32 queries (2 WMMA tiles) per wave
  const int m  = lane & 15;
  const bool hi = (lane & 16) != 0;
  const int off  = hi ? 16 : 0;              // unused (B frags also use base)
  const int base = hi ? 8  : 0;              // A-frag K-dim offset / C-row offset

  // ---- loop-invariant RoPE inverse frequencies (YaRN degenerates: scale==1) ----
  float invfQ[16], invfK[16];
#pragma unroll
  for (int e = 0; e < 16; ++e) {
    invfQ[e] = __powf(10000.f, -(float)(base + (e < 8 ? e : 8 + e)) * (1.f / 32.f));
    const int x = base + (e < 8 ? e : 8 + e);
    invfK[e] = __powf(10000.f, -(float)x * (1.f / 32.f));
  }

  // ---------------- Q fragments (B-layout), RoPE + scale folded in -------------
  v16bf Qf[2][6];
#pragma unroll
  for (int t2 = 0; t2 < 2; ++t2) {
    const int s = qbase + 16 * t2 + m;
    const size_t row = (size_t)(b * Ss + s) * Hh + h;
    const float* qnr = qn + row * DNn;
    const float* qpr = qp + row * DRr;
#pragma unroll
    for (int f = 0; f < 4; ++f) {
      const float* pbase = qnr + 32 * f + base;
#pragma unroll
      for (int q4 = 0; q4 < 4; ++q4) {
        const float4 x = *(const float4*)(pbase + 4 * q4 + (q4 >= 2 ? 8 : 0));
        Qf[t2][f][4 * q4 + 0] = (__bf16)(x.x * QSCALE);
        Qf[t2][f][4 * q4 + 1] = (__bf16)(x.y * QSCALE);
        Qf[t2][f][4 * q4 + 2] = (__bf16)(x.z * QSCALE);
        Qf[t2][f][4 * q4 + 3] = (__bf16)(x.w * QSCALE);
      }
    }
    // f=4 covers dims x (0..31), f=5 covers x+32 — one sincos feeds both
#pragma unroll
    for (int e = 0; e < 16; ++e) {
      const int x = base + (e < 8 ? e : 8 + e);
      float sn, cs; __sincosf((float)s * invfQ[e], &sn, &cs);
      const float a = qpr[x], bb = qpr[x + 32];
      Qf[t2][4][e] = (__bf16)((a * cs - bb * sn) * QSCALE);
      Qf[t2][5][e] = (__bf16)((bb * cs + a * sn) * QSCALE);
    }
  }

  // ---------------- K fragments (A-layout), shared by both query tiles ----------
  auto load_k = [&](int kb, v16bf (&Kf)[6]) {
    int sk = kb + m; if (sk > Ss - 1) sk = Ss - 1;          // clamp (masked anyway)
    const size_t row = (size_t)(b * Ss + sk) * Hh + h;
    const float* knr = kn + row * DNn;
    const float* kpr = kp + (size_t)(b * Ss + sk) * DRr;    // k_pe has H==1
#pragma unroll
    for (int f = 0; f < 4; ++f) {
#pragma unroll
      for (int half = 0; half < 2; ++half) {
        const float* pbase = knr + 32 * f + base + 16 * half;
#pragma unroll
        for (int q4 = 0; q4 < 2; ++q4) {
          const float4 x = *(const float4*)(pbase + 4 * q4);
          const int e = 8 * half + 4 * q4;
          Kf[f][e + 0] = (__bf16)x.x;
          Kf[f][e + 1] = (__bf16)x.y;
          Kf[f][e + 2] = (__bf16)x.z;
          Kf[f][e + 3] = (__bf16)x.w;
        }
      }
    }
#pragma unroll
    for (int e = 0; e < 16; ++e) {
      const int x = base + (e < 8 ? e : 8 + e);             // 0..31
      float sn, cs; __sincosf((float)sk * invfK[e], &sn, &cs);
      const float a = kpr[x], bb = kpr[x + 32];
      Kf[4][e] = (__bf16)(a * cs - bb * sn);
      Kf[5][e] = (__bf16)(bb * cs + a * sn);
    }
  };

  // ---------------- V fragments (B-layout), shared by both query tiles ----------
  auto load_v = [&](int kb, int t) -> v16bf {
    v16bf Vf;
    const int dv = 16 * t + m;
#pragma unroll
    for (int e = 0; e < 16; ++e) {
      int sk = kb + base + (e < 8 ? e : 8 + e); if (sk > Ss - 1) sk = Ss - 1;
      Vf[e] = (__bf16)vv[((size_t)(b * Ss + sk) * Hh + h) * DVv + dv];
    }
    return Vf;
  };

  // ---------------- flash-attention main loop -----------------------------------
  float Mrow[2] = {-__builtin_inff(), -__builtin_inff()};
  float Lrow[2] = {0.f, 0.f};
  v8f Oacc[2][8];
#pragma unroll
  for (int t2 = 0; t2 < 2; ++t2)
#pragma unroll
    for (int t = 0; t < 8; ++t)
#pragma unroll
      for (int r = 0; r < 8; ++r) Oacc[t2][t][r] = 0.f;

  const int kend = qbase + 31;
  for (int kb = 0; kb <= kend; kb += 32) {
    v16bf Kf[6];
    v8f Sacc[2][2];
#pragma unroll
    for (int t2 = 0; t2 < 2; ++t2)
#pragma unroll
      for (int r = 0; r < 8; ++r) { Sacc[t2][0][r] = 0.f; Sacc[t2][1][r] = 0.f; }

    // S^T = K_tile x Q^T for both query tiles (K frags loaded once, used twice)
    load_k(kb, Kf);
#pragma unroll
    for (int f = 0; f < 6; ++f) {
      Sacc[0][0] = WMMA_BF16(Kf[f], Qf[0][f], Sacc[0][0]);
      Sacc[1][0] = WMMA_BF16(Kf[f], Qf[1][f], Sacc[1][0]);
    }
    load_k(kb + 16, Kf);
#pragma unroll
    for (int f = 0; f < 6; ++f) {
      Sacc[0][1] = WMMA_BF16(Kf[f], Qf[0][f], Sacc[0][1]);
      Sacc[1][1] = WMMA_BF16(Kf[f], Qf[1][f], Sacc[1][1]);
    }

    // base-2 online softmax per query tile (key n in VGPR slot, query m in lane)
    v16bf Pf[2];
#pragma unroll
    for (int t2 = 0; t2 < 2; ++t2) {
      const int mabs = qbase + 16 * t2 + m;
      float mx = -__builtin_inff();
#pragma unroll
      for (int r = 0; r < 8; ++r) {
        const int n0 = kb + base + r;
        const float s0 = (n0      <= mabs) ? Sacc[t2][0][r] : -__builtin_inff();
        const float s1 = (n0 + 16 <= mabs) ? Sacc[t2][1][r] : -__builtin_inff();
        Sacc[t2][0][r] = s0; Sacc[t2][1][r] = s1;
        mx = fmaxf(mx, fmaxf(s0, s1));
      }
      mx = fmaxf(mx, __shfl_xor(mx, 16));
      const float Mnew  = fmaxf(Mrow[t2], mx);
      const float alpha = exp2f(Mrow[t2] - Mnew);
      float lsum = 0.f;
#pragma unroll
      for (int r = 0; r < 8; ++r) {
        const float e0 = exp2f(Sacc[t2][0][r] - Mnew);
        const float e1 = exp2f(Sacc[t2][1][r] - Mnew);
        lsum += e0 + e1;
        Pf[t2][r]     = (__bf16)e0;        // lands directly in A-frag layout
        Pf[t2][r + 8] = (__bf16)e1;
      }
      lsum += __shfl_xor(lsum, 16);
      Lrow[t2] = Lrow[t2] * alpha + lsum;
      Mrow[t2] = Mnew;
#pragma unroll
      for (int r = 0; r < 8; ++r) {
        const float aO = __shfl(alpha, base + r);
#pragma unroll
        for (int t = 0; t < 8; ++t) Oacc[t2][t][r] *= aO;
      }
    }

    // O += P x V : each V fragment loaded once, used by both query tiles
#pragma unroll
    for (int t = 0; t < 8; ++t) {
      const v16bf Vf = load_v(kb, t);
      Oacc[0][t] = WMMA_BF16(Pf[0], Vf, Oacc[0][t]);
      Oacc[1][t] = WMMA_BF16(Pf[1], Vf, Oacc[1][t]);
    }
  }

  // ---------------- epilogue: normalize, store O and LSE (log2 units) ----------
#pragma unroll
  for (int t2 = 0; t2 < 2; ++t2) {
#pragma unroll
    for (int r = 0; r < 8; ++r) {
      const float Lr  = __shfl(Lrow[t2], base + r);
      const float inv = 1.0f / Lr;
      const int s = qbase + 16 * t2 + base + r;
      float* orow = o + ((size_t)(b * Ss + s) * Hh + h) * DVv + m;
#pragma unroll
      for (int t = 0; t < 8; ++t) orow[16 * t] = Oacc[t2][t][r] * inv;
    }
    if (lane < 16) {
      const int s = qbase + 16 * t2 + lane;
      lse[(size_t)(b * Ss + s) * Hh + h] = Mrow[t2] + __log2f(Lrow[t2]);
    }
  }
}

extern "C" void kernel_launch(void* const* d_in, const int* in_sizes, int n_in,
                              void* d_out, int out_size, void* d_ws, size_t ws_size,
                              hipStream_t stream) {
  (void)in_sizes; (void)n_in; (void)out_size; (void)d_ws; (void)ws_size;
  const float* qn = (const float*)d_in[0];
  const float* qp = (const float*)d_in[1];
  const float* kn = (const float*)d_in[2];
  const float* kp = (const float*)d_in[3];
  const float* vv = (const float*)d_in[4];
  float* o   = (float*)d_out;
  float* lse = o + (size_t)Bb * Ss * Hh * DVv;
  dim3 grid(Ss / 32, Hh, Bb);
  mla_fwd_kernel<<<grid, dim3(32, 1, 1), 0, stream>>>(qn, qp, kn, kp, vv, o, lse);
}
